// GCN_24068996727451
// MI455X (gfx1250) — matrix operand order repaired and checked by reading, not compiled, not run on hardware
//
#include <hip/hip_runtime.h>
#include <hip/hip_bf16.h>

// ---------------------------------------------------------------------------
// GCN on gfx1250: edge scatter-add (L2-resident atomics) + WMMA f16 GEMMs.
// ---------------------------------------------------------------------------

#define N_NODES  10000
#define N_EDGES  640000
#define N_GRAPHS 64
#define FEAT     128
#define M_PAD    10016   // 626 * 16, padded row count for 16x16 WMMA tiling

typedef __attribute__((ext_vector_type(16))) _Float16 v16h;
typedef __attribute__((ext_vector_type(8)))  _Float16 v8h;
typedef __attribute__((ext_vector_type(8)))  float    v8f;

// ---------------- utility: zero a float region (grid-stride) ----------------
__global__ void gcn_zero_f32(float* __restrict__ p, int n) {
    int i = blockIdx.x * blockDim.x + threadIdx.x;
    int stride = gridDim.x * blockDim.x;
    for (; i < n; i += stride) p[i] = 0.0f;
}

// ---------------- degree accumulation over edges ----------------------------
__global__ void gcn_degree(const int* __restrict__ src, const int* __restrict__ dst,
                           float* __restrict__ out_deg, float* __restrict__ in_deg) {
    int i = blockIdx.x * blockDim.x + threadIdx.x;
    int stride = gridDim.x * blockDim.x;
    for (; i < N_EDGES; i += stride) {
        atomicAdd(&out_deg[src[i]], 1.0f);
        atomicAdd(&in_deg[dst[i]], 1.0f);
    }
}

// ---------------- norm = deg^{-1/2} (deg==0 -> 1) ---------------------------
__global__ void gcn_norm(const float* __restrict__ od, const float* __restrict__ id,
                         float* __restrict__ ns, float* __restrict__ nd) {
    int i = blockIdx.x * blockDim.x + threadIdx.x;
    if (i < N_NODES) {
        float o = od[i];
        float d = id[i];
        ns[i] = rsqrtf(o > 0.0f ? o : 1.0f);
        nd[i] = rsqrtf(d > 0.0f ? d : 1.0f);
    }
}

// ---------------- W (fp32 row-major [K,N]) -> Wt (f16 col-major [N,K]) ------
__global__ void gcn_wconv(const float* __restrict__ W, _Float16* __restrict__ Wt) {
    int i = blockIdx.x * blockDim.x + threadIdx.x;  // i = n*FEAT + k
    if (i < FEAT * FEAT) {
        int n = i >> 7;
        int k = i & 127;
        Wt[i] = (_Float16)W[k * FEAT + n];
    }
}

// ---------------- edge scatter: agg[dst] += x[src] * norm_src[src] ----------
// One wave per edge; lane covers 4 features (32*4 = 128). Index/norm loads
// happen once per wave. Prefetch next edge's source row into cache.
__global__ void gcn_scatter(const float* __restrict__ x, const int* __restrict__ src,
                            const int* __restrict__ dst, const float* __restrict__ ns,
                            float* __restrict__ agg) {
    int gtid   = blockIdx.x * blockDim.x + threadIdx.x;
    int wave   = gtid >> 5;
    int lane   = threadIdx.x & 31;
    int nwaves = (gridDim.x * blockDim.x) >> 5;
    for (int e = wave; e < N_EDGES; e += nwaves) {
        int s = src[e];
        int d = dst[e];
        int e2 = e + nwaves;
        if (e2 < N_EDGES) {
            __builtin_prefetch(x + (size_t)src[e2] * FEAT + lane * 4, 0, 0);
        }
        float w = ns[s];
        const float4 v = *(const float4*)(x + (size_t)s * FEAT + lane * 4);
        float* ap = agg + (size_t)d * FEAT + lane * 4;
        atomicAdd(ap + 0, v.x * w);
        atomicAdd(ap + 1, v.y * w);
        atomicAdd(ap + 2, v.z * w);
        atomicAdd(ap + 3, v.w * w);
    }
}

// ---------------- A16[m,k] = f16(agg[m,k] * norm_dst[m]); pad rows = 0 ------
__global__ void gcn_a16(const float* __restrict__ agg, const float* __restrict__ nd,
                        _Float16* __restrict__ A) {
    int i = blockIdx.x * blockDim.x + threadIdx.x;  // over M_PAD*FEAT
    if (i < M_PAD * FEAT) {
        int m = i >> 7;
        float v = (m < N_NODES) ? agg[i] * nd[m] : 0.0f;
        A[i] = (_Float16)v;
    }
}

// ---------------- WMMA GEMM: out = relu(A @ Bt^T + bias) --------------------
// A: [M_PAD, 128] f16 row-major. Bt: [128, 128] f16 column-major (Bt[n*128+k]).
// Block = 256 threads = 8 waves; wave w owns N-tile w; block owns one M-tile.
// Output buffer is padded to M_PAD rows, so the epilogue stores are
// unconditional (no exec-mask diamonds); pad rows are never read downstream.
__global__ void __launch_bounds__(256)
gcn_gemm(const _Float16* __restrict__ A, const _Float16* __restrict__ Bt,
         const float* __restrict__ bias, float* __restrict__ out) {
    const int mtile = blockIdx.x;          // 0..625
    const int ntile = threadIdx.x >> 5;    // 0..7
    const int lane  = threadIdx.x & 31;
    const int half  = lane >> 4;           // which 16-lane half
    const int idx   = lane & 15;           // M row (for A) / N col (for B)

    const _Float16* arow = A  + (size_t)(mtile * 16 + idx) * FEAT;
    const _Float16* bcol = Bt + (size_t)(ntile * 16 + idx) * FEAT;

    v8f acc = {};
#pragma unroll
    for (int kt = 0; kt < FEAT; kt += 32) {
        // A fragment: elems 0..15 = K(kt + half*16 + 0..15)
        v8h alo = *(const v8h*)(arow + kt + half * 16);
        v8h ahi = *(const v8h*)(arow + kt + half * 16 + 8);
        v16h a = __builtin_shufflevector(alo, ahi,
                                         0, 1, 2, 3, 4, 5, 6, 7,
                                         8, 9, 10, 11, 12, 13, 14, 15);
        // B fragment: lanes 0-15 cover K(kt+0..15), lanes 16-31 K(kt+16..31)
        v16h b = *(const v16h*)(bcol + kt + half * 16);
        acc = __builtin_amdgcn_wmma_f32_16x16x32_f16(
            /*neg_a=*/false, a, /*neg_b=*/false, b,
            /*c_mod=*/(short)0, acc, /*reuse_a=*/false, /*reuse_b=*/false);
    }

    const int n  = ntile * 16 + idx;
    const float bn = bias[n];
    float* orow = out + (size_t)(mtile * 16 + half * 8) * FEAT + n;
#pragma unroll
    for (int r = 0; r < 8; ++r) {
        // C layout: VGPR r -> M = mtile*16 + half*8 + r, N = n
        float v = acc[r] + bn;
        orow[(size_t)r * FEAT] = v > 0.0f ? v : 0.0f;
    }
}

// ---------------- per-graph pooling (sum + count) ---------------------------
__global__ void gcn_pool(const float* __restrict__ h, const int* __restrict__ gid,
                         float* __restrict__ pooled, float* __restrict__ counts) {
    int i = blockIdx.x * blockDim.x + threadIdx.x;  // over N_NODES*FEAT
    if (i < N_NODES * FEAT) {
        int m = i >> 7;
        int f = i & 127;
        int g = gid[m];
        atomicAdd(&pooled[g * FEAT + f], h[i]);
        if (f == 0) atomicAdd(&counts[g], 1.0f);
    }
}

// ---------------- final GEMV: out[g] = mean(pooled[g]) . Wd + bd ------------
__global__ void gcn_final(const float* __restrict__ pooled, const float* __restrict__ counts,
                          const float* __restrict__ Wd, const float* __restrict__ bd,
                          float* __restrict__ out) {
    int g = threadIdx.x;
    if (g < N_GRAPHS) {
        float c = counts[g];
        c = c > 1.0f ? c : 1.0f;
        float s = 0.0f;
        for (int f = 0; f < FEAT; ++f) s += pooled[g * FEAT + f] * Wd[f];
        out[g] = s / c + bd[0];
    }
}

// ---------------------------------------------------------------------------
extern "C" void kernel_launch(void* const* d_in, const int* in_sizes, int n_in,
                              void* d_out, int out_size, void* d_ws, size_t ws_size,
                              hipStream_t stream) {
    (void)in_sizes; (void)n_in; (void)out_size; (void)ws_size;

    const float* in_feat = (const float*)d_in[0];
    const int*   src     = (const int*)d_in[1];
    const int*   dst     = (const int*)d_in[2];
    const int*   gid     = (const int*)d_in[3];
    const float* W1      = (const float*)d_in[4];
    const float* b1      = (const float*)d_in[5];
    const float* W2      = (const float*)d_in[6];
    const float* b2      = (const float*)d_in[7];
    const float* Wd      = (const float*)d_in[8];
    const float* bd      = (const float*)d_in[9];
    float* out = (float*)d_out;

    // ---- workspace layout (floats) ----
    float* ws      = (float*)d_ws;
    float* out_deg = ws;                               // 10016
    float* in_deg  = out_deg + M_PAD;                  // 10016
    float* norm_s  = in_deg + M_PAD;                   // 10016
    float* norm_d  = norm_s + M_PAD;                   // 10016
    float* counts  = norm_d + M_PAD;                   // 64
    float* pooled  = counts + 64;                      // 64*128
    float* agg     = pooled + N_GRAPHS * FEAT;         // M_PAD*128
    float* hbuf    = agg + (size_t)M_PAD * FEAT;       // M_PAD*128 (padded rows writable)
    _Float16* A16  = (_Float16*)(hbuf + (size_t)M_PAD * FEAT);  // M_PAD*128 halfs
    _Float16* W1t  = A16 + (size_t)M_PAD * FEAT;       // 128*128 halfs
    _Float16* W2t  = W1t + FEAT * FEAT;                // 128*128 halfs

    const int ZB = 256;
    const int n_small  = 4 * M_PAD + 64 + N_GRAPHS * FEAT;  // deg/norm/counts/pooled
    const int n_agg    = M_PAD * FEAT;
    const int n_nodefe = N_NODES * FEAT;

    // ---- zero scratch (degrees/norms/counts/pooled + agg) ----
    gcn_zero_f32<<<(n_small + ZB - 1) / ZB, ZB, 0, stream>>>(ws, n_small);
    gcn_zero_f32<<<(n_agg + ZB - 1) / ZB, ZB, 0, stream>>>(agg, n_agg);

    // ---- degrees and norms ----
    gcn_degree<<<2500, 256, 0, stream>>>(src, dst, out_deg, in_deg);
    gcn_norm<<<(N_NODES + 255) / 256, 256, 0, stream>>>(out_deg, in_deg, norm_s, norm_d);

    // ---- weight conversion (fp32 [K,N] -> f16 col-major [N,K]) ----
    gcn_wconv<<<(FEAT * FEAT + 255) / 256, 256, 0, stream>>>(W1, W1t);
    gcn_wconv<<<(FEAT * FEAT + 255) / 256, 256, 0, stream>>>(W2, W2t);

    // ---- layer 1: scatter -> f16 convert -> WMMA GEMM + bias + relu ----
    gcn_scatter<<<2048, 256, 0, stream>>>(in_feat, src, dst, norm_s, agg);
    gcn_a16<<<(n_agg + 255) / 256, 256, 0, stream>>>(agg, norm_d, A16);
    gcn_gemm<<<M_PAD / 16, 256, 0, stream>>>(A16, W1t, b1, hbuf);

    // ---- layer 2 ----
    gcn_zero_f32<<<(n_agg + ZB - 1) / ZB, ZB, 0, stream>>>(agg, n_agg);
    gcn_scatter<<<2048, 256, 0, stream>>>(hbuf, src, dst, norm_s, agg);
    gcn_a16<<<(n_agg + 255) / 256, 256, 0, stream>>>(agg, norm_d, A16);
    gcn_gemm<<<M_PAD / 16, 256, 0, stream>>>(A16, W2t, b2, hbuf);

    // ---- per-graph mean pool + final linear ----
    gcn_pool<<<(n_nodefe + 255) / 256, 256, 0, stream>>>(hbuf, gid, pooled, counts);
    gcn_final<<<1, 64, 0, stream>>>(pooled, counts, Wd, bd, out);
}
